// MultiHeadAttention_76699525972695
// MI455X (gfx1250) — matrix operand order repaired and checked by reading, not compiled, not run on hardware
//
#include <hip/hip_runtime.h>
#include <hip/hip_bf16.h>

// ---------------------------------------------------------------------------
// Types for CDNA5 WMMA bf16 16x16x32:  D(f32 16x16) = A(bf16 16x32) * B(bf16 32x16) + C
// ---------------------------------------------------------------------------
typedef __attribute__((ext_vector_type(16))) __bf16 bf16x16;
typedef __attribute__((ext_vector_type(8)))  float  f32x8;

union Frag {           // 16 bf16 per lane = 8 VGPRs = two 16B chunks
    bf16x16 v;
    uint4   u[2];
};

__device__ __forceinline__ f32x8 wmma_bf16(const Frag& a, const Frag& b, f32x8 c) {
    // 8 args: (neg_a, A, neg_b, B, c_mod, C, reuse_a, reuse_b)
    return __builtin_amdgcn_wmma_f32_16x16x32_bf16(false, a.v, false, b.v,
                                                   (short)0, c, false, false);
}

__device__ __forceinline__ unsigned short f2bf(float f) {
    unsigned int u = __float_as_uint(f);
    u += 0x7fffu + ((u >> 16) & 1u);   // round-to-nearest-even
    return (unsigned short)(u >> 16);
}

#define NEG_MASK (-1e9f)

// ---------------------------------------------------------------------------
// fp32 -> bf16 elementwise convert
// ---------------------------------------------------------------------------
__global__ void convert_f32_bf16(const float* __restrict__ in,
                                 unsigned short* __restrict__ out, int n) {
    int i = blockIdx.x * blockDim.x + threadIdx.x;
    int stride = gridDim.x * blockDim.x;
    for (; i < n; i += stride) out[i] = f2bf(in[i]);
}

// ---------------------------------------------------------------------------
// W[k][n] fp32  ->  WT[n][k] bf16   (LDS tiled transpose, n = 1024)
// ---------------------------------------------------------------------------
__global__ void transpose_to_bf16(const float* __restrict__ W,
                                  unsigned short* __restrict__ WT, int n) {
    __shared__ float tile[32][33];
    const int tx = threadIdx.x;
    const int xin = blockIdx.x * 32 + tx;          // n index on input
    const int y0  = blockIdx.y * 32;               // k base
    for (int i = threadIdx.y; i < 32; i += 8)
        tile[i][tx] = W[(size_t)(y0 + i) * n + xin];
    __syncthreads();
    const int xout = blockIdx.y * 32 + tx;         // k index on output
    for (int i = threadIdx.y; i < 32; i += 8)
        WT[(size_t)(blockIdx.x * 32 + i) * n + xout] = f2bf(tile[tx][i]);
}

// ---------------------------------------------------------------------------
// Generic WMMA GEMM:  out[M,N] = (A[M,K] * BT[N,K]^T + bias) * scale
//   mode 0: out -> bf16 row-major [M,N]
//   mode 1: out -> bf16 Vt layout  [b][n][s]  (b = row>>11, s = row&2047)
//   mode 2: out -> fp32 row-major [M,N]
// Each wave computes a 64x64 tile (4x4 WMMA frags); block = 8 waves = 128x256.
// ---------------------------------------------------------------------------
__global__ __launch_bounds__(256) void gemm_bf16w(
        const unsigned short* __restrict__ A,
        const unsigned short* __restrict__ BT,
        const float* __restrict__ bias, float scale,
        unsigned short* __restrict__ outb, float* __restrict__ outf,
        int M, int N, int K, int mode) {
    const int lane = threadIdx.x & 31;
    const int wave = threadIdx.x >> 5;
    const int half = lane >> 4;
    const int l16  = lane & 15;
    const int m0 = blockIdx.x * 128 + (wave >> 2) * 64;
    const int n0 = blockIdx.y * 256 + (wave & 3) * 64;

    f32x8 zero8 = {};
    f32x8 acc[4][4];
#pragma unroll
    for (int i = 0; i < 4; ++i)
#pragma unroll
        for (int j = 0; j < 4; ++j) acc[i][j] = zero8;

    const unsigned short* aBase[4];
    const unsigned short* bBase[4];
#pragma unroll
    for (int i = 0; i < 4; ++i) aBase[i] = A  + (size_t)(m0 + i * 16 + l16) * K;
#pragma unroll
    for (int j = 0; j < 4; ++j) bBase[j] = BT + (size_t)(n0 + j * 16 + l16) * K;

    for (int kk = 0; kk < K; kk += 32) {
        Frag af[4], bf[4];
#pragma unroll
        for (int i = 0; i < 4; ++i) {          // A: e->K = (e/8)*16 + half*8 + e%8
            af[i].u[0] = *(const uint4*)(aBase[i] + kk + half * 8);
            af[i].u[1] = *(const uint4*)(aBase[i] + kk + 16 + half * 8);
        }
#pragma unroll
        for (int j = 0; j < 4; ++j) {          // B: e->K = (e/8)*16 + half*8 + e%8
            bf[j].u[0] = *(const uint4*)(bBase[j] + kk + half * 8);
            bf[j].u[1] = *(const uint4*)(bBase[j] + kk + 16 + half * 8);
        }
#pragma unroll
        for (int i = 0; i < 4; ++i)
#pragma unroll
            for (int j = 0; j < 4; ++j)
                acc[i][j] = wmma_bf16(af[i], bf[j], acc[i][j]);
    }

    // Epilogue: C layout VGPR r -> row = r + 8*half, col = l16
#pragma unroll
    for (int j = 0; j < 4; ++j) {
        const int col = n0 + j * 16 + l16;
        const float bv = bias ? bias[col] : 0.0f;
#pragma unroll
        for (int i = 0; i < 4; ++i) {
            const int rbase = m0 + i * 16 + 8 * half;
#pragma unroll
            for (int r = 0; r < 8; ++r) {
                const int row = rbase + r;
                const float v = (acc[i][j][r] + bv) * scale;
                if (mode == 0) {
                    outb[(size_t)row * N + col] = f2bf(v);
                } else if (mode == 1) {
                    const int bb = row >> 11;
                    const int s  = row & 2047;
                    outb[(size_t)bb * (1024u * 2048u) + (size_t)col * 2048u + s] = f2bf(v);
                } else {
                    outf[(size_t)row * N + col] = v;
                }
            }
        }
    }
}

// ---------------------------------------------------------------------------
// Flash attention (causal), one wave per (b, h, 16-row q tile).
//   Qg/Kg : bf16 [B,S,1024] (pre-scaled by hd^-0.25)
//   Vt    : bf16 [B][h*64+d][S]   (per-head transposed V)
//   Og    : bf16 [B,S,1024]
// ---------------------------------------------------------------------------
__global__ __launch_bounds__(256) void flash_attn(
        const unsigned short* __restrict__ Qg,
        const unsigned short* __restrict__ Kg,
        const unsigned short* __restrict__ Vt,
        unsigned short* __restrict__ Og) {
    __shared__ __align__(16) unsigned short Pl[8][16 * 32];  // per-wave P staging

    const int lane = threadIdx.x & 31;
    const int wave = threadIdx.x >> 5;
    const int half = lane >> 4;
    const int l16  = lane & 15;

    const int id = blockIdx.x * 8 + wave;    // 4096 waves total
    const int qt = id & 127;
    const int h  = (id >> 7) & 15;
    const int b  = id >> 11;
    const int qbase = qt * 16;

    // Q fragments for both 32-wide d-steps (kept in registers all loop)
    const unsigned short* qrow = Qg + (size_t)(b * 2048 + qbase + l16) * 1024 + h * 64;
    Frag qf[2];
#pragma unroll
    for (int d = 0; d < 2; ++d) {
        qf[d].u[0] = *(const uint4*)(qrow + d * 32 + half * 8);
        qf[d].u[1] = *(const uint4*)(qrow + d * 32 + 16 + half * 8);
    }

    f32x8 zero8 = {};
    f32x8 oacc[4];
#pragma unroll
    for (int dt = 0; dt < 4; ++dt) oacc[dt] = zero8;
    float mst[8], lst[8];
#pragma unroll
    for (int r = 0; r < 8; ++r) { mst[r] = -1e30f; lst[r] = 0.0f; }

    unsigned short* pw = &Pl[wave][0];
    const unsigned short* kbase = Kg + (size_t)(b * 2048) * 1024 + h * 64;
    const unsigned short* vbase = Vt + (size_t)b * (1024u * 2048u) + (size_t)(h * 64) * 2048u;

    for (int kb = 0; kb <= qbase; kb += 32) {
        // --- scores: S = Q K^T for two 16-key tiles ---
        Frag kf[2][2];
#pragma unroll
        for (int t = 0; t < 2; ++t)
#pragma unroll
            for (int d = 0; d < 2; ++d) {   // B frag of K^T: contiguous along hd
                const unsigned short* kr =
                    kbase + (size_t)(kb + t * 16 + l16) * 1024 + d * 32;
                kf[t][d].u[0] = *(const uint4*)(kr + half * 8);
                kf[t][d].u[1] = *(const uint4*)(kr + 16 + half * 8);
            }
        f32x8 s0 = zero8, s1 = zero8;
        s0 = wmma_bf16(qf[0], kf[0][0], s0);
        s0 = wmma_bf16(qf[1], kf[0][1], s0);
        s1 = wmma_bf16(qf[0], kf[1][0], s1);
        s1 = wmma_bf16(qf[1], kf[1][1], s1);

        // --- causal mask (only near the diagonal) ---
        if (kb + 31 > qbase) {
#pragma unroll
            for (int r = 0; r < 8; ++r) {
                const int row = qbase + r + 8 * half;
                if (kb + l16 > row)       s0[r] = NEG_MASK;
                if (kb + 16 + l16 > row)  s1[r] = NEG_MASK;
            }
        }

        // --- online softmax; rows live in (VGPR r, lane-half) ---
#pragma unroll
        for (int r = 0; r < 8; ++r) {
            float rm = fmaxf(s0[r], s1[r]);
#pragma unroll
            for (int m = 1; m < 16; m <<= 1) rm = fmaxf(rm, __shfl_xor(rm, m, 32));
            const float mnew  = fmaxf(mst[r], rm);
            const float alpha = __expf(mst[r] - mnew);
            mst[r] = mnew;
            const float p0 = __expf(s0[r] - mnew);
            const float p1 = __expf(s1[r] - mnew);
            float rs = p0 + p1;
#pragma unroll
            for (int m = 1; m < 16; m <<= 1) rs += __shfl_xor(rs, m, 32);
            lst[r] = lst[r] * alpha + rs;
#pragma unroll
            for (int dt = 0; dt < 4; ++dt) oacc[dt][r] *= alpha;
            // stage P (C layout -> LDS row-major [16][32])
            const int rl = r + 8 * half;
            pw[rl * 32 + l16]      = f2bf(p0);
            pw[rl * 32 + 16 + l16] = f2bf(p1);
        }

        // --- re-fragment P as WMMA A (wave-private LDS; DS ops are in-order) ---
        Frag pf;
        pf.u[0] = *(const uint4*)(pw + l16 * 32 + half * 8);
        pf.u[1] = *(const uint4*)(pw + l16 * 32 + 16 + half * 8);

        // prefetch next K tile row for this lane
        if (kb + 32 <= qbase)
            __builtin_prefetch(kbase + (size_t)(kb + 32 + l16) * 1024, 0, 1);

        // --- O += P * V, per 16-wide d tile (Vt gives contiguous B frags) ---
#pragma unroll
        for (int dt = 0; dt < 4; ++dt) {
            const unsigned short* vr = vbase + (size_t)(dt * 16 + l16) * 2048u + kb;
            Frag vf;
            vf.u[0] = *(const uint4*)(vr + half * 8);
            vf.u[1] = *(const uint4*)(vr + 16 + half * 8);
            oacc[dt] = wmma_bf16(pf, vf, oacc[dt]);
        }
    }

    // --- normalize and write O (bf16, [B,S,1024]) ---
#pragma unroll
    for (int r = 0; r < 8; ++r) {
        const float inv = 1.0f / lst[r];
        const int row = qbase + r + 8 * half;
        unsigned short* op = Og + (size_t)(b * 2048 + row) * 1024 + h * 64 + l16;
#pragma unroll
        for (int dt = 0; dt < 4; ++dt) op[dt * 16] = f2bf(oacc[dt][r] * inv);
    }
}

// ---------------------------------------------------------------------------
// Host-side launch. Inputs: x, mask, Wq, bq, Wk, Wv, bv, Wo, bo (mask unused:
// causal masking is applied analytically with the same -1e9 constant).
// ---------------------------------------------------------------------------
extern "C" void kernel_launch(void* const* d_in, const int* in_sizes, int n_in,
                              void* d_out, int out_size, void* d_ws, size_t ws_size,
                              hipStream_t stream) {
    const float* x  = (const float*)d_in[0];
    const float* Wq = (const float*)d_in[2];
    const float* bq = (const float*)d_in[3];
    const float* Wk = (const float*)d_in[4];
    const float* Wv = (const float*)d_in[5];
    const float* bv = (const float*)d_in[6];
    const float* Wo = (const float*)d_in[7];
    const float* bo = (const float*)d_in[8];
    float* out = (float*)d_out;

    // Workspace layout (bf16 elements): 48 MB total
    unsigned short* ws  = (unsigned short*)d_ws;
    unsigned short* xb  = ws;                   // [4096,1024]
    unsigned short* WqT = xb  + 4194304;        // [1024,1024] transposed
    unsigned short* WkT = WqT + 1048576;
    unsigned short* WvT = WkT + 1048576;
    unsigned short* WoT = WvT + 1048576;
    unsigned short* Qg  = WoT + 1048576;        // [4096,1024]
    unsigned short* Kg  = Qg  + 4194304;
    unsigned short* Vt  = Kg  + 4194304;        // [B][1024][2048]
    unsigned short* Og  = Vt  + 4194304;        // [4096,1024]

    convert_f32_bf16<<<4096, 256, 0, stream>>>(x, xb, 4096 * 1024);

    dim3 tb(32, 8), tg(32, 32);
    transpose_to_bf16<<<tg, tb, 0, stream>>>(Wq, WqT, 1024);
    transpose_to_bf16<<<tg, tb, 0, stream>>>(Wk, WkT, 1024);
    transpose_to_bf16<<<tg, tb, 0, stream>>>(Wv, WvT, 1024);
    transpose_to_bf16<<<tg, tb, 0, stream>>>(Wo, WoT, 1024);

    const float scale = 0.35355339059327379f;   // 64^-0.25, applied to q and k
    dim3 gg(4096 / 128, 1024 / 256), gb(256);
    gemm_bf16w<<<gg, gb, 0, stream>>>(xb, WqT, bq, scale, Qg, nullptr, 4096, 1024, 1024, 0);
    gemm_bf16w<<<gg, gb, 0, stream>>>(xb, WkT, nullptr, scale, Kg, nullptr, 4096, 1024, 1024, 0);
    gemm_bf16w<<<gg, gb, 0, stream>>>(xb, WvT, bv, 1.0f, Vt, nullptr, 4096, 1024, 1024, 1);

    flash_attn<<<512, 256, 0, stream>>>(Qg, Kg, Vt, Og);

    gemm_bf16w<<<gg, gb, 0, stream>>>(Og, WoT, bo, 1.0f, nullptr, out, 4096, 1024, 1024, 2);
}
